// GAT_54331336294862
// MI455X (gfx1250) — matrix operand order repaired and checked by reading, not compiled, not run on hardware
//
#include <hip/hip_runtime.h>
#include <stdint.h>

#define NGRAPH 64
#define EPSLN 1e-5f
#define NEG_SLOPE 0.2f

typedef __attribute__((ext_vector_type(16))) __bf16 v16bf;
typedef __attribute__((ext_vector_type(8)))  float  v8f;
typedef __attribute__((ext_vector_type(4)))  unsigned int v4u;

// types for the async-to-LDS builtin: int vector_size(16), global/LDS pointers
typedef int v4i32 __attribute__((vector_size(16)));
typedef __attribute__((address_space(1))) v4i32* v4i32_gptr;
typedef __attribute__((address_space(3))) v4i32* v4i32_lptr;

union BFrag { v16bf v; v4u q[2]; };

__device__ __forceinline__ unsigned short f32_to_bf16(float f) {
  union { float f; unsigned u; } x; x.f = f;
  unsigned r = x.u + 0x7FFFu + ((x.u >> 16) & 1u);   // round-to-nearest-even
  return (unsigned short)(r >> 16);
}

// ---------------- conversion kernels ----------------
__global__ void cvt_x_kernel(const float* __restrict__ x, unsigned short* __restrict__ xb,
                             long long n) {
  long long t = (long long)blockIdx.x * blockDim.x + threadIdx.x;
  if (t < n) xb[t] = f32_to_bf16(x[t]);
}

// wT[j*256+i] = bf16(W[i*256+j])  (store W transposed: row = output col, contiguous K)
__global__ void cvt_wT_kernel(const float* __restrict__ w, unsigned short* __restrict__ wT) {
  int t = blockIdx.x * blockDim.x + threadIdx.x;   // 65536 threads
  int i = t >> 8, j = t & 255;
  wT[j * 256 + i] = f32_to_bf16(w[t]);
}

// ---------------- WMMA GEMM: C[N,256] = A[N,256] * W[256,256] (bf16 in, f32 out) ---------
// Block tile 128x64 (8 waves x 16 rows), W tile staged in LDS once per block.
#define LDS_STRIDE 264   // 256 + 8 halfword pad -> ds_load_b128 hits all 64 banks
__global__ __launch_bounds__(256) void gemm_bf16_wmma(
    const unsigned short* __restrict__ A,   // [N,256] bf16 row-major
    const unsigned short* __restrict__ BT,  // [256,256] bf16, BT[n][k] = W[k][n]
    float* __restrict__ C, int N) {
  __shared__ unsigned short lb[64 * LDS_STRIDE];   // 64 cols x 256 K (padded), ~33 KB
  const int lane = threadIdx.x & 31;
  const int wave = threadIdx.x >> 5;
  const int lm   = lane & 15;
  const int lh   = lane >> 4;
  const int col0 = blockIdx.y * 64;

  // ---- stage 64 columns of W^T (32 KB) into LDS: 2048 16-byte chunks / 256 threads ----
  #pragma unroll
  for (int i = 0; i < 8; ++i) {
    int t  = i * 256 + threadIdx.x;
    int r  = t >> 5;                 // column within tile (0..63)
    int cj = (t & 31) * 8;           // K offset in halfwords (16B granules)
    const unsigned short* gsrc = BT + (size_t)(col0 + r) * 256 + cj;
    unsigned short* ldst = lb + r * LDS_STRIDE + cj;
#if defined(__has_builtin) && __has_builtin(__builtin_amdgcn_global_load_async_to_lds_b128)
    __builtin_amdgcn_global_load_async_to_lds_b128(
        (v4i32_gptr)(v4i32*)(gsrc),
        (v4i32_lptr)(v4i32*)(ldst), 0, 0);
#else
    *(v4u*)ldst = *(const v4u*)gsrc;
#endif
  }
#if defined(__has_builtin) && __has_builtin(__builtin_amdgcn_global_load_async_to_lds_b128)
#if __has_builtin(__builtin_amdgcn_s_wait_asynccnt)
  __builtin_amdgcn_s_wait_asynccnt(0);
#else
  asm volatile("s_wait_asynccnt 0x0" ::: "memory");
#endif
#endif
  __syncthreads();

  int row0 = blockIdx.x * 128 + wave * 16;
  int rowA = row0 + lm; if (rowA > N - 1) rowA = N - 1;   // branchless clamp, EXEC stays all-1
  // A layout (16-bit 16x32): lane<16 holds K {0..7,16..23}; lane>=16 holds {8..15,24..31}
  const unsigned short* ap0 = A + (size_t)rowA * 256 + lh * 8;
  // B layout (32x16): lane = column, lane<16 K {0..7,16..23}; lane>=16 K {8..15,24..31}
  const unsigned short* lb0 = lb + lm * LDS_STRIDE + lh * 8;
  v8f acc0 = {}; v8f acc1 = {}; v8f acc2 = {}; v8f acc3 = {};
  #pragma unroll
  for (int k = 0; k < 256; k += 32) {
    BFrag a, b0, b1, b2, b3;
    a.q[0] = *(const v4u*)(ap0 + k);
    a.q[1] = *(const v4u*)(ap0 + k + 16);
    const unsigned short* bp = lb0 + k;
    b0.q[0] = *(const v4u*)(bp);                      b0.q[1] = *(const v4u*)(bp + 16);
    b1.q[0] = *(const v4u*)(bp + 16 * LDS_STRIDE);    b1.q[1] = *(const v4u*)(bp + 16 * LDS_STRIDE + 16);
    b2.q[0] = *(const v4u*)(bp + 32 * LDS_STRIDE);    b2.q[1] = *(const v4u*)(bp + 32 * LDS_STRIDE + 16);
    b3.q[0] = *(const v4u*)(bp + 48 * LDS_STRIDE);    b3.q[1] = *(const v4u*)(bp + 48 * LDS_STRIDE + 16);
    acc0 = __builtin_amdgcn_wmma_f32_16x16x32_bf16(false, a.v, false, b0.v, (short)0, acc0, false, false);
    acc1 = __builtin_amdgcn_wmma_f32_16x16x32_bf16(false, a.v, false, b1.v, (short)0, acc1, false, false);
    acc2 = __builtin_amdgcn_wmma_f32_16x16x32_bf16(false, a.v, false, b2.v, (short)0, acc2, false, false);
    acc3 = __builtin_amdgcn_wmma_f32_16x16x32_bf16(false, a.v, false, b3.v, (short)0, acc3, false, false);
  }
  // C/D layout: vgpr r -> M = r + (lane>=16 ? 8 : 0), N = lane&15
  int rowB = row0 + lh * 8;
  int colC = col0 + lm;
  #pragma unroll
  for (int r = 0; r < 8; ++r) {
    int m = rowB + r;
    if (m < N) {
      float* cp = C + (size_t)m * 256 + colC;
      cp[0]  = acc0[r];
      cp[16] = acc1[r];
      cp[32] = acc2[r];
      cp[48] = acc3[r];
    }
  }
}

// ---------------- attention scalar dots + init ----------------
__global__ void attn_kernel(const float* __restrict__ h,
                            const float* __restrict__ a_src,
                            const float* __restrict__ a_dst,
                            float* __restrict__ als, float* __restrict__ ald,
                            float* __restrict__ mx, float* __restrict__ dn,
                            int N, int H, int C) {
  int t = blockIdx.x * blockDim.x + threadIdx.x;
  if (t >= N * H) return;
  int n = t / H, hh = t - n * H;
  const float* hp = h + (size_t)n * 256 + hh * C;
  const float* as = a_src + hh * C;
  const float* ad = a_dst + hh * C;
  float ss = 0.f, sd = 0.f;
  for (int c = 0; c < C; ++c) { float v = hp[c]; ss += v * as[c]; sd += v * ad[c]; }
  als[t] = ss; ald[t] = sd;
  mx[t] = -3.402823466e38f;   // -FLT_MAX
  dn[t] = 0.f;
}

// ---------------- edge passes ----------------
__device__ __forceinline__ void edge_ends(const int* __restrict__ ei, int E, int e,
                                          int& s, int& d) {
  if (e < E) { s = ei[e]; d = ei[E + e]; } else { s = e - E; d = s; }  // self loops appended
}

__device__ __forceinline__ void atomic_max_f32(float* addr, float v) {
  if (v >= 0.f) atomicMax((int*)addr, __float_as_int(v));
  else          atomicMin((unsigned int*)addr, __float_as_uint(v));
}

__global__ void edge_max_kernel(const int* __restrict__ ei,
                                const float* __restrict__ als, const float* __restrict__ ald,
                                float* __restrict__ mx, int E, int ET, int H) {
  long long t = (long long)blockIdx.x * blockDim.x + threadIdx.x;
  if (t >= (long long)ET * H) return;
  int e = (int)(t / H), hh = (int)(t % H);
  int s, d; edge_ends(ei, E, e, s, d);
  float v = als[s * H + hh] + ald[d * H + hh];
  v = v > 0.f ? v : NEG_SLOPE * v;
  atomic_max_f32(&mx[d * H + hh], v);
}

__global__ void edge_exp_kernel(const int* __restrict__ ei,
                                const float* __restrict__ als, const float* __restrict__ ald,
                                const float* __restrict__ mx, float* __restrict__ dn,
                                float* __restrict__ ex, int E, int ET, int H) {
  long long t = (long long)blockIdx.x * blockDim.x + threadIdx.x;
  if (t >= (long long)ET * H) return;
  int e = (int)(t / H), hh = (int)(t % H);
  int s, d; edge_ends(ei, E, e, s, d);
  float v = als[s * H + hh] + ald[d * H + hh];
  v = v > 0.f ? v : NEG_SLOPE * v;
  float z = __expf(v - mx[d * H + hh]);
  ex[t] = z;
  atomicAdd(&dn[d * H + hh], z);
}

// 64 threads per edge, float4 per thread: b128 gather + 4x global_atomic_add_f32 scatter
__global__ __launch_bounds__(256) void edge_msg_kernel(
    const int* __restrict__ ei, const float* __restrict__ h,
    const float* __restrict__ ex, const float* __restrict__ dn,
    float* __restrict__ acc, int E, int ET, int H, int logC) {
  int e = blockIdx.x * 4 + (threadIdx.x >> 6);
  if (e >= ET) return;
  int c = (threadIdx.x & 63) * 4;
  int s, d; edge_ends(ei, E, e, s, d);
  int hh = c >> logC;                        // logC >= 6, so all 4 features share a head
  float alpha = ex[(size_t)e * H + hh] / dn[d * H + hh];
  const float4 hv = *(const float4*)(h + (size_t)s * 256 + c);
  float* ap = acc + (size_t)d * 256 + c;
  atomicAdd(ap + 0, alpha * hv.x);
  atomicAdd(ap + 1, alpha * hv.y);
  atomicAdd(ap + 2, alpha * hv.z);
  atomicAdd(ap + 3, alpha * hv.w);
}

// ---------------- bias + layernorm + ELU (one wave32 per node) ----------------
__global__ __launch_bounds__(256) void ln_elu_kernel(
    float* __restrict__ acc, const float* __restrict__ bias,
    const float* __restrict__ g, const float* __restrict__ be,
    unsigned short* __restrict__ xb, int N, int writeBf) {
  int wave = threadIdx.x >> 5;
  int lane = threadIdx.x & 31;
  int n = blockIdx.x * 8 + wave;
  if (n >= N) return;
  float* row = acc + (size_t)n * 256;
  int f0 = lane * 8;
  float v[8];
  float s = 0.f;
  #pragma unroll
  for (int i = 0; i < 8; ++i) { v[i] = row[f0 + i] + bias[f0 + i]; s += v[i]; }
  #pragma unroll
  for (int m = 16; m > 0; m >>= 1) s += __shfl_xor(s, m, 32);
  float mu = s * (1.f / 256.f);
  float q = 0.f;
  #pragma unroll
  for (int i = 0; i < 8; ++i) { float dv = v[i] - mu; q += dv * dv; }
  #pragma unroll
  for (int m = 16; m > 0; m >>= 1) q += __shfl_xor(q, m, 32);
  float inv = rsqrtf(q * (1.f / 256.f) + EPSLN);
  #pragma unroll
  for (int i = 0; i < 8; ++i) {
    float y = (v[i] - mu) * inv * g[f0 + i] + be[f0 + i];
    y = y > 0.f ? y : (__expf(y) - 1.f);                   // ELU
    row[f0 + i] = y;
    if (writeBf) xb[(size_t)n * 256 + f0 + i] = f32_to_bf16(y);
  }
}

// ---------------- pooling + FC ----------------
__global__ void pool_kernel(const float* __restrict__ x, const int* __restrict__ batch,
                            float* __restrict__ pooled, float* __restrict__ cnt, int N) {
  int n = blockIdx.x; int c = threadIdx.x;
  int gidx = batch[n];
  atomicAdd(&pooled[gidx * 256 + c], x[(size_t)n * 256 + c]);
  if (c == 0) atomicAdd(&cnt[gidx], 1.f);
}

__global__ void fc_kernel(const float* __restrict__ pooled, const float* __restrict__ cnt,
                          const float* __restrict__ fcw, const float* __restrict__ fcb,
                          float* __restrict__ out) {
  int gidx = blockIdx.x; int j = threadIdx.x;
  float cg = cnt[gidx]; cg = cg > 1.f ? cg : 1.f;
  float s = 0.f;
  for (int i = 0; i < 256; ++i) s += pooled[gidx * 256 + i] * fcw[i * 256 + j];
  s = s / cg + fcb[j];
  out[gidx * 256 + j] = s > 0.f ? s : 0.f;
}

// ---------------- driver ----------------
extern "C" void kernel_launch(void* const* d_in, const int* in_sizes, int n_in,
                              void* d_out, int out_size, void* d_ws, size_t ws_size,
                              hipStream_t stream) {
  const float* x0    = (const float*)d_in[0];
  const int*   ei    = (const int*)d_in[1];
  const int*   batch = (const int*)d_in[2];
  int N  = in_sizes[0] / 256;
  int E  = in_sizes[1] / 2;
  int ET = E + N;

  char* ws = (char*)d_ws;
  size_t off = 0;
  auto carve = [&](size_t bytes) -> char* {
    char* p = ws + off; off = (off + bytes + 255) & ~(size_t)255; return p;
  };
  unsigned short* xb = (unsigned short*)carve((size_t)N * 256 * 2);
  unsigned short* wT = (unsigned short*)carve((size_t)256 * 256 * 2);
  float* h      = (float*)carve((size_t)N * 256 * 4);
  float* acc    = (float*)carve((size_t)N * 256 * 4);
  float* als    = (float*)carve((size_t)N * 4 * 4);
  float* ald    = (float*)carve((size_t)N * 4 * 4);
  float* mx     = (float*)carve((size_t)N * 4 * 4);
  float* dn     = (float*)carve((size_t)N * 4 * 4);
  float* ex     = (float*)carve((size_t)ET * 4 * 4);
  float* pooled = (float*)carve((size_t)NGRAPH * 256 * 4);
  float* cnt    = (float*)carve((size_t)NGRAPH * 4);
  (void)ws_size; (void)n_in; (void)out_size;

  {
    long long tot = (long long)N * 256;
    cvt_x_kernel<<<(int)((tot + 255) / 256), 256, 0, stream>>>(x0, xb, tot);
  }

  dim3 gGemm((N + 127) / 128, 4);
  for (int l = 0; l < 3; ++l) {
    const float* W  = (const float*)d_in[3 + 6 * l];
    const float* as = (const float*)d_in[4 + 6 * l];
    const float* ad = (const float*)d_in[5 + 6 * l];
    const float* bi = (const float*)d_in[6 + 6 * l];
    const float* ga = (const float*)d_in[7 + 6 * l];
    const float* be = (const float*)d_in[8 + 6 * l];
    int H    = (l == 2) ? 1 : 4;
    int C    = 256 / H;
    int logC = (l == 2) ? 8 : 6;

    cvt_wT_kernel<<<256, 256, 0, stream>>>(W, wT);
    gemm_bf16_wmma<<<gGemm, 256, 0, stream>>>(xb, wT, h, N);

    attn_kernel<<<(N * H + 255) / 256, 256, 0, stream>>>(h, as, ad, als, ald, mx, dn, N, H, C);
    (void)hipMemsetAsync(acc, 0, (size_t)N * 256 * 4, stream);
    {
      long long tot = (long long)ET * H;
      int blocks = (int)((tot + 255) / 256);
      edge_max_kernel<<<blocks, 256, 0, stream>>>(ei, als, ald, mx, E, ET, H);
      edge_exp_kernel<<<blocks, 256, 0, stream>>>(ei, als, ald, mx, dn, ex, E, ET, H);
    }
    edge_msg_kernel<<<(ET + 3) / 4, 256, 0, stream>>>(ei, h, ex, dn, acc, E, ET, H, logC);
    ln_elu_kernel<<<(N + 7) / 8, 256, 0, stream>>>(acc, bi, ga, be, xb, N, (l < 2) ? 1 : 0);
  }

  (void)hipMemsetAsync(pooled, 0, (size_t)NGRAPH * 256 * 4, stream);
  (void)hipMemsetAsync(cnt, 0, (size_t)NGRAPH * 4, stream);
  pool_kernel<<<N, 256, 0, stream>>>(acc, batch, pooled, cnt, N);
  fc_kernel<<<NGRAPH, 256, 0, stream>>>(pooled, cnt,
                                        (const float*)d_in[21], (const float*)d_in[22],
                                        (float*)d_out);
}
